// Social_LSTM_abs_11295763989012
// MI455X (gfx1250) — matrix operand order repaired and checked by reading, not compiled, not run on hardware
//
#include <hip/hip_runtime.h>
#include <hip/hip_bf16.h>
#include <math.h>

typedef __attribute__((ext_vector_type(16))) _Float16 v16h;
typedef __attribute__((ext_vector_type(8)))  _Float16 v8h;
typedef __attribute__((ext_vector_type(8)))  float    v8f;

#define N_AG   1024
#define E_DIM  64
#define H_DIM  128
#define G_DIM  8
#define G2     64
#define K_SOC  (G2 * H_DIM)     /* 8192 */
#define K_CAT  256              /* 2E + H */
#define GATES  512              /* 4H */
#define OUT_PAD 128
#define OUT_DIM 120
#define NEIGH_SZ 0.4f
#define KSPLIT 4
#define KSLICE (K_SOC / KSPLIT) /* 2048 */

#define WMMA_F16(a, b, c) \
    __builtin_amdgcn_wmma_f32_16x16x32_f16(false, (a), false, (b), (short)0, (c), false, false)

// ---------------------------------------------------------------------------
// Fragment loaders (wave32 WMMA; this file's own k order, same for A and B)
// A 16x32 f16: lane group hl holds K = {16hl..16hl+15} contiguous
// B 32x16 f16: lane group hl holds K = {16hl..16hl+15} contiguous
// ---------------------------------------------------------------------------
__device__ __forceinline__ v16h load_a_frag(const _Float16* p) {
    return *(const v16h*)(p);
}
__device__ __forceinline__ v16h load_b_frag(const _Float16* p) {
    return *(const v16h*)(p);
}

// ---------------------------------------------------------------------------
// Prep kernels
// ---------------------------------------------------------------------------
__global__ void k_cvt_f16(const float* __restrict__ src, _Float16* __restrict__ dst, int n) {
    int i = blockIdx.x * blockDim.x + threadIdx.x;
    if (i < n) dst[i] = (_Float16)src[i];
}

// Wcat[e, k] : k<128 -> W_ih[e,k], else W_hh[e,k-128]   (512 x 256)
__global__ void k_build_wcat(const float* __restrict__ wih, const float* __restrict__ whh,
                             _Float16* __restrict__ wcat) {
    int i = blockIdx.x * blockDim.x + threadIdx.x;
    if (i >= GATES * K_CAT) return;
    int e = i >> 8, k = i & 255;
    float v = (k < 128) ? wih[e * 128 + k] : whh[e * 128 + (k - 128)];
    wcat[i] = (_Float16)v;
}

// Wout padded to 128x128 (rows >= 120 zero)
__global__ void k_build_wout(const float* __restrict__ wout, _Float16* __restrict__ dst) {
    int i = blockIdx.x * blockDim.x + threadIdx.x;
    if (i >= OUT_PAD * H_DIM) return;
    int e = i >> 7, k = i & 127;
    dst[i] = (e < OUT_DIM) ? (_Float16)wout[e * H_DIM + k] : (_Float16)0.0f;
}

// emb_x = relu(xabs @ W_emb^T + b_emb) -> act[:, 0:64] (f16)
__global__ void k_emb(const float* __restrict__ x, const float* __restrict__ W,
                      const float* __restrict__ b, _Float16* __restrict__ act) {
    int i = blockIdx.x * blockDim.x + threadIdx.x;
    if (i >= N_AG * E_DIM) return;
    int n = i >> 6, e = i & 63;
    float v = x[2 * n] * W[2 * e] + x[2 * n + 1] * W[2 * e + 1] + b[e];
    v = v > 0.0f ? v : 0.0f;
    act[n * K_CAT + e] = (_Float16)v;
}

// h0 -> act[:, 128:256] (f16)
__global__ void k_h0(const float* __restrict__ h0, _Float16* __restrict__ act) {
    int i = blockIdx.x * blockDim.x + threadIdx.x;
    if (i >= N_AG * H_DIM) return;
    int n = i >> 7, h = i & 127;
    act[n * K_CAT + 2 * E_DIM + h] = (_Float16)h0[i];
}

// ---------------------------------------------------------------------------
// Social pooling: one block (128 threads) per agent n.
// LDS accumulator 64 cells x 128 = 32KB, plus per-neighbor cell table.
// One-hot mask exploited as a scatter-add (vs 17 GFLOP dense masked einsum).
// ---------------------------------------------------------------------------
__global__ void __launch_bounds__(128)
k_social(const float* __restrict__ xabs, const float* __restrict__ h0,
         _Float16* __restrict__ social) {
    __shared__ float acc[G2 * H_DIM];
    __shared__ int   cellArr[N_AG];
    const int n = blockIdx.x;
    const int tid = threadIdx.x;

    for (int i = tid; i < G2 * H_DIM; i += 128) acc[i] = 0.0f;

    const float xn = xabs[2 * n], yn = xabs[2 * n + 1];
    for (int m = tid; m < N_AG; m += 128) {
        float dx = xabs[2 * m]     - (xn - NEIGH_SZ * 0.5f);
        float dy = xabs[2 * m + 1] - (yn - NEIGH_SZ * 0.5f);
        int cx = (int)floorf(dx / NEIGH_SZ * (float)G_DIM);
        int cy = (int)floorf(dy / NEIGH_SZ * (float)G_DIM);
        bool valid = (dx >= 0.0f) && (dx < NEIGH_SZ) && (dy >= 0.0f) && (dy < NEIGH_SZ)
                  && (cx >= 0) && (cx < G_DIM) && (cy >= 0) && (cy < G_DIM) && (m != n);
        cellArr[m] = valid ? (cx + cy * G_DIM) : -1;
    }
    __syncthreads();

    // Thread tid owns hidden column tid; branch is block-uniform per m.
    for (int m = 0; m < N_AG; ++m) {
        int c = cellArr[m];
        if (c >= 0) acc[c * H_DIM + tid] += h0[m * H_DIM + tid];
    }
    __syncthreads();

    _Float16* row = social + (size_t)n * K_SOC;
    for (int g = 0; g < G2; ++g)
        row[g * H_DIM + tid] = (_Float16)acc[g * H_DIM + tid];
}

// ---------------------------------------------------------------------------
// GEMM 1: pooling partials = social @ Wsoc^T   [1024x8192]x[8192x64]
// - 16 rows x 64 cols per wave (4 accumulators): one A-frag feeds 4 WMMAs.
// - K split 4 ways (256 waves); deterministic reduce epilogue.
// - Explicit register double-buffering: next iteration's fragments are loaded
//   BEFORE the current WMMAs, so waits are partial and loads overlap WMMA.
// ---------------------------------------------------------------------------
__global__ void __launch_bounds__(256)
k_pool_gemm(const _Float16* __restrict__ A, const _Float16* __restrict__ B,
            float* __restrict__ part) {
    int wave = (blockIdx.x * blockDim.x + threadIdx.x) >> 5;
    int lane = threadIdx.x & 31;
    int ks = wave >> 6;                 // 0..3 K slice
    int tm = wave & 63;                 // 0..63 M tile
    int hl = lane >> 4, r = lane & 15;

    const _Float16* pa  = A + (size_t)(tm * 16 + r) * K_SOC + ks * KSLICE + hl * 16;
    const _Float16* pb0 = B + (size_t)r * K_SOC + ks * KSLICE + hl * 16;
    const _Float16* pb1 = pb0 + (size_t)16 * K_SOC;
    const _Float16* pb2 = pb1 + (size_t)16 * K_SOC;
    const _Float16* pb3 = pb2 + (size_t)16 * K_SOC;

    v8f c0 = {}, c1 = {}, c2 = {}, c3 = {};

    // Prologue: stage k=0 fragments.
    v16h a  = load_a_frag(pa);
    v16h b0 = load_b_frag(pb0);
    v16h b1 = load_b_frag(pb1);
    v16h b2 = load_b_frag(pb2);
    v16h b3 = load_b_frag(pb3);

    for (int k = 32; k < KSLICE; k += 32) {
        __builtin_prefetch(pa + k + 512, 0, 1);       // global_prefetch_b8
        // Issue next iteration's loads first (separate regs -> partial waits).
        v16h an  = load_a_frag(pa  + k);
        v16h bn0 = load_b_frag(pb0 + k);
        v16h bn1 = load_b_frag(pb1 + k);
        v16h bn2 = load_b_frag(pb2 + k);
        v16h bn3 = load_b_frag(pb3 + k);
        // Consume current fragments (loaded one iteration ago).
        c0 = WMMA_F16(a, b0, c0);
        c1 = WMMA_F16(a, b1, c1);
        c2 = WMMA_F16(a, b2, c2);
        c3 = WMMA_F16(a, b3, c3);
        a = an; b0 = bn0; b1 = bn1; b2 = bn2; b3 = bn3;
    }
    // Tail.
    c0 = WMMA_F16(a, b0, c0);
    c1 = WMMA_F16(a, b1, c1);
    c2 = WMMA_F16(a, b2, c2);
    c3 = WMMA_F16(a, b3, c3);

    // part[ks][row][col], row = tm*16 + hl*8 + j, col = {0,16,32,48} + r
    float* base = part + ((size_t)ks * N_AG + tm * 16 + hl * 8) * E_DIM + r;
    #pragma unroll
    for (int j = 0; j < 8; ++j) {
        float* rowp = base + (size_t)j * E_DIM;
        rowp[0]  = c0[j];
        rowp[16] = c1[j];
        rowp[32] = c2[j];
        rowp[48] = c3[j];
    }
}

// reduce K-slice partials + bias + relu -> act[:, 64:128] (f16)
__global__ void k_pool_reduce(const float* __restrict__ part, const float* __restrict__ bias,
                              _Float16* __restrict__ act) {
    int i = blockIdx.x * blockDim.x + threadIdx.x;
    if (i >= N_AG * E_DIM) return;
    float v = part[i] + part[i + N_AG * E_DIM] + part[i + 2 * N_AG * E_DIM]
            + part[i + 3 * N_AG * E_DIM] + bias[i & 63];
    v = v > 0.0f ? v : 0.0f;
    act[(i >> 6) * K_CAT + E_DIM + (i & 63)] = (_Float16)v;
}

// ---------------------------------------------------------------------------
// GEMM 2: gates = act @ Wcat^T + b_ih + b_hh   [1024x256]x[256x512] -> f32
// 16x64 per wave: 64 M-tiles x 8 col-groups = 512 waves. Fully unrolled K.
// ---------------------------------------------------------------------------
__global__ void __launch_bounds__(256)
k_gates_gemm(const _Float16* __restrict__ A, const _Float16* __restrict__ B,
             const float* __restrict__ bi, const float* __restrict__ bh,
             float* __restrict__ gates) {
    int wave = (blockIdx.x * blockDim.x + threadIdx.x) >> 5;
    int lane = threadIdx.x & 31;
    int tm = wave >> 3, ng = wave & 7;  // 64 x 8(col-groups of 64)
    int hl = lane >> 4, r = lane & 15;

    const _Float16* pa  = A + (size_t)(tm * 16 + r) * K_CAT + hl * 16;
    const _Float16* pb0 = B + (size_t)(ng * 64 + r) * K_CAT + hl * 16;
    const _Float16* pb1 = pb0 + (size_t)16 * K_CAT;
    const _Float16* pb2 = pb1 + (size_t)16 * K_CAT;
    const _Float16* pb3 = pb2 + (size_t)16 * K_CAT;

    v8f c0 = {}, c1 = {}, c2 = {}, c3 = {};
    #pragma unroll
    for (int k = 0; k < K_CAT; k += 32) {
        v16h a = load_a_frag(pa + k);
        c0 = WMMA_F16(a, load_b_frag(pb0 + k), c0);
        c1 = WMMA_F16(a, load_b_frag(pb1 + k), c1);
        c2 = WMMA_F16(a, load_b_frag(pb2 + k), c2);
        c3 = WMMA_F16(a, load_b_frag(pb3 + k), c3);
    }

    int col0 = ng * 64 + r;
    float b0 = bi[col0]      + bh[col0];
    float b1 = bi[col0 + 16] + bh[col0 + 16];
    float b2 = bi[col0 + 32] + bh[col0 + 32];
    float b3 = bi[col0 + 48] + bh[col0 + 48];
    float* base = gates + (size_t)(tm * 16 + hl * 8) * GATES + col0;
    #pragma unroll
    for (int j = 0; j < 8; ++j) {
        float* rowp = base + (size_t)j * GATES;
        rowp[0]  = c0[j] + b0;
        rowp[16] = c1[j] + b1;
        rowp[32] = c2[j] + b2;
        rowp[48] = c3[j] + b3;
    }
}

// ---------------------------------------------------------------------------
// LSTM elementwise (torch gate order i,f,g,o)
// ---------------------------------------------------------------------------
__global__ void k_lstm(const float* __restrict__ gates, const float* __restrict__ c0,
                       _Float16* __restrict__ hout) {
    int idx = blockIdx.x * blockDim.x + threadIdx.x;
    if (idx >= N_AG * H_DIM) return;
    int n = idx >> 7, h = idx & 127;
    const float* g = gates + n * GATES;
    float ig = g[h], fg = g[128 + h], gg = g[256 + h], og = g[384 + h];
    float si = 1.0f / (1.0f + expf(-ig));
    float sf = 1.0f / (1.0f + expf(-fg));
    float so = 1.0f / (1.0f + expf(-og));
    float cc = sf * c0[idx] + si * tanhf(gg);
    float hv = so * tanhf(cc);
    hout[idx] = (_Float16)hv;
}

// ---------------------------------------------------------------------------
// GEMM 3: final = h @ Wout^T + b_out, scattered into the 6-way split output.
// 16x64 per wave: 64 M-tiles x 2 col-groups = 128 waves. Fully unrolled K.
// out[j6*20480 + n*20 + t] = final[n, j6*20 + t]
// ---------------------------------------------------------------------------
__global__ void __launch_bounds__(256)
k_final_gemm(const _Float16* __restrict__ A, const _Float16* __restrict__ B,
             const float* __restrict__ bias, float* __restrict__ out) {
    int wave = (blockIdx.x * blockDim.x + threadIdx.x) >> 5;
    int lane = threadIdx.x & 31;
    int tm = wave >> 1, ng = wave & 1;  // 64 x 2(col-groups of 64)
    int hl = lane >> 4, r = lane & 15;

    const _Float16* pa  = A + (size_t)(tm * 16 + r) * H_DIM + hl * 16;
    const _Float16* pb0 = B + (size_t)(ng * 64 + r) * H_DIM + hl * 16;
    const _Float16* pb1 = pb0 + (size_t)16 * H_DIM;
    const _Float16* pb2 = pb1 + (size_t)16 * H_DIM;
    const _Float16* pb3 = pb2 + (size_t)16 * H_DIM;

    v8f c0 = {}, c1 = {}, c2 = {}, c3 = {};
    #pragma unroll
    for (int k = 0; k < H_DIM; k += 32) {
        v16h a = load_a_frag(pa + k);
        c0 = WMMA_F16(a, load_b_frag(pb0 + k), c0);
        c1 = WMMA_F16(a, load_b_frag(pb1 + k), c1);
        c2 = WMMA_F16(a, load_b_frag(pb2 + k), c2);
        c3 = WMMA_F16(a, load_b_frag(pb3 + k), c3);
    }

    v8f cs[4] = {c0, c1, c2, c3};
    #pragma unroll
    for (int t4 = 0; t4 < 4; ++t4) {
        int col = ng * 64 + t4 * 16 + r;
        if (col < OUT_DIM) {
            int j6 = col / 20;
            int t  = col - j6 * 20;
            float bv = bias[col];
            #pragma unroll
            for (int j = 0; j < 8; ++j) {
                int row = tm * 16 + hl * 8 + j;
                out[j6 * (N_AG * 20) + row * 20 + t] = cs[t4][j] + bv;
            }
        }
    }
}

// ---------------------------------------------------------------------------
// Host launcher
// ---------------------------------------------------------------------------
extern "C" void kernel_launch(void* const* d_in, const int* in_sizes, int n_in,
                              void* d_out, int out_size, void* d_ws, size_t ws_size,
                              hipStream_t stream) {
    const float* xabs  = (const float*)d_in[0];
    const float* i_h0  = (const float*)d_in[1];
    const float* i_c0  = (const float*)d_in[2];
    const float* W_emb = (const float*)d_in[3];
    const float* b_emb = (const float*)d_in[4];
    const float* W_soc = (const float*)d_in[5];
    const float* b_soc = (const float*)d_in[6];
    const float* W_ih  = (const float*)d_in[7];
    const float* W_hh  = (const float*)d_in[8];
    const float* b_ih  = (const float*)d_in[9];
    const float* b_hh  = (const float*)d_in[10];
    const float* W_out = (const float*)d_in[11];
    const float* b_out = (const float*)d_in[12];
    float* out = (float*)d_out;

    char* ws = (char*)d_ws;
    _Float16* social   = (_Float16*)(ws);                        // 16 MB   [1024 x 8192]
    _Float16* wsoc16   = (_Float16*)(ws + 16777216);             // 1 MB    [64 x 8192]
    _Float16* act16    = (_Float16*)(ws + 17825792);             // 512 KB  [1024 x 256]
    _Float16* wcat16   = (_Float16*)(ws + 18350080);             // 256 KB  [512 x 256]
    _Float16* wout16   = (_Float16*)(ws + 18612224);             // 32 KB   [128 x 128]
    float*    gates    = (float*)   (ws + 18644992);             // 2 MB    [1024 x 512]
    _Float16* hbuf     = (_Float16*)(ws + 20742144);             // 256 KB  [1024 x 128]
    float*    poolpart = (float*)   (ws + 21004288);             // 1 MB    [4 x 1024 x 64]

    // --- prep ---
    k_cvt_f16   <<<(E_DIM * K_SOC + 255) / 256, 256, 0, stream>>>(W_soc, wsoc16, E_DIM * K_SOC);
    k_build_wcat<<<(GATES * K_CAT + 255) / 256, 256, 0, stream>>>(W_ih, W_hh, wcat16);
    k_build_wout<<<(OUT_PAD * H_DIM + 255) / 256, 256, 0, stream>>>(W_out, wout16);
    k_emb       <<<(N_AG * E_DIM + 255) / 256, 256, 0, stream>>>(xabs, W_emb, b_emb, act16);
    k_h0        <<<(N_AG * H_DIM + 255) / 256, 256, 0, stream>>>(i_h0, act16);

    // --- social scatter-pool ---
    k_social<<<N_AG, 128, 0, stream>>>(xabs, i_h0, social);

    // --- pooling GEMM: 4 K-slices x 64 M-tiles = 256 waves -> 32 blocks ---
    k_pool_gemm<<<32, 256, 0, stream>>>(social, wsoc16, poolpart);
    k_pool_reduce<<<(N_AG * E_DIM + 255) / 256, 256, 0, stream>>>(poolpart, b_soc, act16);

    // --- gates GEMM: 512 waves -> 64 blocks ---
    k_gates_gemm<<<64, 256, 0, stream>>>(act16, wcat16, b_ih, b_hh, gates);

    // --- LSTM elementwise ---
    k_lstm<<<(N_AG * H_DIM + 255) / 256, 256, 0, stream>>>(gates, i_c0, hbuf);

    // --- output GEMM + split scatter: 128 waves -> 16 blocks ---
    k_final_gemm<<<16, 256, 0, stream>>>(hbuf, wout16, b_out, out);
}
